// CausalSelfAttention_64398739636831
// MI455X (gfx1250) — matrix operand order repaired and checked by reading, not compiled, not run on hardware
//
#include <hip/hip_runtime.h>
#include <stdint.h>

// ---------------------------------------------------------------------------
// CDNA5 (gfx1250, wave32) causal self-attention.
// - All matmuls: v_wmma_f32_16x16x32_bf16 (f32 accumulate).
// - GEMMs: 32x32 tile per wave (4 WMMA per 8 b128 loads).
// - Attention: flash-style streaming softmax; K/V tiles staged to LDS with
//   global_load_async_to_lds_b128 (ASYNCcnt) shared by all 4 waves of a block.
// ---------------------------------------------------------------------------

typedef __attribute__((ext_vector_type(16))) __bf16 v16bf;
typedef __attribute__((ext_vector_type(8)))  float  v8f;

#define SEQ 4096
#define DM  768
#define NH  12
#define HD  64

// fp32 -> bf16, round-to-nearest-even
__device__ __forceinline__ unsigned short f2bf(float x) {
  unsigned int u = __float_as_uint(x);
  u += 0x7FFFu + ((u >> 16) & 1u);
  return (unsigned short)(u >> 16);
}

union Frag16 { v16bf v; uint4 u[2]; };

// A-fragment (16x32 bf16, M x K), row-major source, stride in elements.
// lanes 0-15: M=lane, K=0..7 / 16..23 ; lanes 16-31: M=lane-16, K=8..15 / 24..31
__device__ __forceinline__ v16bf load_a16(const unsigned short* base, int rowStride, int lane) {
  int hsel = lane >> 4;
  int m    = lane & 15;
  const unsigned short* p = base + (size_t)m * rowStride + hsel * 8;
  Frag16 f;
  f.u[0] = *(const uint4*)(p);
  f.u[1] = *(const uint4*)(p + 16);
  return f.v;
}

// B-fragment (32x16 bf16, K x N); source row n contiguous in K, stride in elements.
// lanes 0-15: N=lane, K=0..7 / 16..23 ; lanes 16-31: N=lane-16, K=8..15 / 24..31
__device__ __forceinline__ v16bf load_b16(const unsigned short* base, int rowStride, int lane) {
  int hsel = lane >> 4;
  int n    = lane & 15;
  const unsigned short* p = base + (size_t)n * rowStride + hsel * 8;
  Frag16 f;
  f.u[0] = *(const uint4*)(p);
  f.u[1] = *(const uint4*)(p + 16);
  return f.v;
}

#define WMMA_BF16(a, b, c) \
  __builtin_amdgcn_wmma_f32_16x16x32_bf16(false, (a), false, (b), (short)0, (c), false, false)

// Async 16B copy global -> LDS (ASYNCcnt path). ldsByteOff = addr[31:0] of an
// LDS-aperture generic pointer (ISA 10.2).
__device__ __forceinline__ void async_ld16(unsigned int ldsByteOff, const void* gaddr) {
  asm volatile("global_load_async_to_lds_b128 %0, %1, off"
               :: "v"(ldsByteOff), "v"(gaddr) : "memory");
}
__device__ __forceinline__ unsigned int lds_off(const void* p) {
  return (unsigned int)(uintptr_t)p;
}

// ---------------------------------------------------------------------------
// fp32 -> bf16 conversion pass
// ---------------------------------------------------------------------------
__global__ void __launch_bounds__(256) cvt_f32_bf16(const float* __restrict__ in,
                                                    unsigned short* __restrict__ out, int n) {
  int i = blockIdx.x * 256 + threadIdx.x;
  if (i < n) out[i] = f2bf(in[i]);
}

// ---------------------------------------------------------------------------
// Fused QKV projection: [4096x768] x [2304x768]^T.  Wave = 32x32 output tile.
// Q, K written [h][s][d]; V written transposed [h][d][s] for the PV stage.
// ---------------------------------------------------------------------------
__global__ void __launch_bounds__(128) qkv_kernel(const unsigned short* __restrict__ xb,
                                                  const unsigned short* __restrict__ Wq,
                                                  const unsigned short* __restrict__ Wk,
                                                  const unsigned short* __restrict__ Wv,
                                                  unsigned short* __restrict__ Qb,
                                                  unsigned short* __restrict__ Kb,
                                                  unsigned short* __restrict__ Vt) {
  int lane = threadIdx.x & 31;
  int wid  = threadIdx.x >> 5;
  int tile = blockIdx.x * 4 + wid;           // (4096/32)*(2304/32) = 128*72 tiles
  int mt = tile / 72;
  int nt = tile % 72;
  int m0  = mt << 5;
  int n0g = nt << 5;
  int sel  = n0g / DM;                       // 0=Q, 1=K, 2=V (768%32==0: no straddle)
  int nloc = n0g % DM;
  const unsigned short* W = (sel == 0) ? Wq : ((sel == 1) ? Wk : Wv);

  const unsigned short* a0p = xb + (size_t)m0 * DM;
  const unsigned short* a1p = a0p + (size_t)16 * DM;
  const unsigned short* b0p = W + (size_t)nloc * DM;
  const unsigned short* b1p = b0p + (size_t)16 * DM;

  v8f c00 = {}, c01 = {}, c10 = {}, c11 = {};
  for (int k = 0; k < DM; k += 32) {
    __builtin_prefetch(a0p + k + 32, 0, 3);
    v16bf a0 = load_a16(a0p + k, DM, lane);
    v16bf a1 = load_a16(a1p + k, DM, lane);
    v16bf b0 = load_b16(b0p + k, DM, lane);
    v16bf b1 = load_b16(b1p + k, DM, lane);
    c00 = WMMA_BF16(a0, b0, c00);
    c01 = WMMA_BF16(a0, b1, c01);
    c10 = WMMA_BF16(a1, b0, c10);
    c11 = WMMA_BF16(a1, b1, c11);
  }

  int hsel = lane >> 4;
  int nC   = lane & 15;
#pragma unroll
  for (int mi = 0; mi < 2; ++mi) {
#pragma unroll
    for (int ni = 0; ni < 2; ++ni) {
      v8f c = (mi == 0) ? ((ni == 0) ? c00 : c01) : ((ni == 0) ? c10 : c11);
      int col = nloc + ni * 16 + nC;
      int h = col >> 6;
      int d = col & 63;
#pragma unroll
      for (int r = 0; r < 8; ++r) {
        int mrow = m0 + mi * 16 + r + 8 * hsel;
        unsigned short v = f2bf(c[r]);
        if (sel == 0)      Qb[((size_t)h * SEQ + mrow) * HD + d] = v;
        else if (sel == 1) Kb[((size_t)h * SEQ + mrow) * HD + d] = v;
        else               Vt[((size_t)h * HD + d) * SEQ + mrow] = v;
      }
    }
  }
}

// ---------------------------------------------------------------------------
// Flash attention. Block = 4 waves = one (head, 64-query-row) tile; wave w owns
// rows [q0b + 16w, q0b + 16w + 16). Per 32-key chunk the block stages
// K (32x64) and V^T (64x32) tiles into LDS with async b128 copies, then each
// wave does 4 WMMA (QK^T) + online softmax + 4 WMMA (P*V) from LDS.
// All waves run the block-uniform trip count; causal masking makes the extra
// chunks exact no-ops for low-row waves.
// ---------------------------------------------------------------------------
__global__ void __launch_bounds__(128) attn_kernel(const unsigned short* __restrict__ Qb,
                                                   const unsigned short* __restrict__ Kb,
                                                   const unsigned short* __restrict__ Vt,
                                                   unsigned short* __restrict__ Yb) {
  __shared__ __align__(16) unsigned short Kl[32 * 64];        // [key][dim]
  __shared__ __align__(16) unsigned short Vl[64 * 32];        // [dim][key]
  __shared__ __align__(16) unsigned short pbuf[4][16 * 32];   // per-wave P relayout

  int tIdx = threadIdx.x;
  int lane = tIdx & 31;
  int wid  = tIdx >> 5;
  int blk  = blockIdx.x;                     // 12 heads * 64 q-blocks
  int h    = blk >> 6;
  int q0b  = (blk & 63) << 6;
  int q0   = q0b + wid * 16;
  int hsel = lane >> 4;
  int nC   = lane & 15;

  // Q tile fragments (K-dim 0..31, 32..63), loaded once from global
  const unsigned short* qbase = Qb + ((size_t)h * SEQ + q0) * HD;
  v16bf aq0 = load_a16(qbase,      HD, lane);
  v16bf aq1 = load_a16(qbase + 32, HD, lane);

  v8f o0 = {}, o1 = {}, o2 = {}, o3 = {};
  float m[8], l[8];
#pragma unroll
  for (int r = 0; r < 8; ++r) { m[r] = -1e30f; l[r] = 0.0f; }

  const float scale = 0.125f;                // 1/sqrt(64)
  unsigned short* pb = pbuf[wid];

  // Cooperative async-stage address split (per thread: 2+2 b128 copies)
  int kr = tIdx >> 2, kc = tIdx & 3;         // K tile: 32 rows x 8 16B-groups
  int vr = tIdx >> 1, vc = tIdx & 1;         // V tile: 64 rows x 4 16B-groups
  unsigned int lk = lds_off(Kl + kr * 64 + kc * 8);
  unsigned int lv = lds_off(Vl + vr * 32 + vc * 8);

  const int kend = q0b + 64;                 // uniform trip count for the block
  for (int key0 = 0; key0 < kend; key0 += 32) {
    // ---- stage K/V chunk into LDS (ASYNCcnt) ----
    const unsigned short* gk = Kb + ((size_t)h * SEQ + key0 + kr) * HD + kc * 8;
    const unsigned short* gv = Vt + ((size_t)h * HD + vr) * SEQ + key0 + vc * 8;
    async_ld16(lk,      gk);                 // K cols kc*8 .. +7
    async_ld16(lk + 64, gk + 32);            // K cols kc*8+32 .. +39
    async_ld16(lv,      gv);                 // V keys vc*8 .. +7
    async_ld16(lv + 32, gv + 16);            // V keys vc*8+16 .. +23
    asm volatile("s_wait_asynccnt 0" ::: "memory");
    __syncthreads();

    // ---- S = Q K^T from LDS ----
    v8f c0 = {}, c1 = {};
    c0 = WMMA_BF16(aq0, load_b16(Kl,            64, lane), c0);
    c0 = WMMA_BF16(aq1, load_b16(Kl + 32,       64, lane), c0);
    c1 = WMMA_BF16(aq0, load_b16(Kl + 16 * 64,      64, lane), c1);
    c1 = WMMA_BF16(aq1, load_b16(Kl + 16 * 64 + 32, 64, lane), c1);

    // ---- online softmax in C-fragment layout ----
#pragma unroll
    for (int r = 0; r < 8; ++r) {
      int M = r + 8 * hsel;
      int q = q0 + M;
      float x0 = c0[r] * scale; if (key0 + nC      > q) x0 = -1e30f;
      float x1 = c1[r] * scale; if (key0 + 16 + nC > q) x1 = -1e30f;
      float t = fmaxf(x0, x1);               // xor 1/2/4/8 stays within 16-lane half
      t = fmaxf(t, __shfl_xor(t, 1, 32));
      t = fmaxf(t, __shfl_xor(t, 2, 32));
      t = fmaxf(t, __shfl_xor(t, 4, 32));
      t = fmaxf(t, __shfl_xor(t, 8, 32));
      float mn    = fmaxf(m[r], t);
      float alpha = __expf(m[r] - mn);
      float p0 = __expf(x0 - mn);
      float p1 = __expf(x1 - mn);
      float rs = p0 + p1;
      rs += __shfl_xor(rs, 1, 32);
      rs += __shfl_xor(rs, 2, 32);
      rs += __shfl_xor(rs, 4, 32);
      rs += __shfl_xor(rs, 8, 32);
      l[r] = l[r] * alpha + rs;
      m[r] = mn;
      o0[r] *= alpha; o1[r] *= alpha; o2[r] *= alpha; o3[r] *= alpha;
      pb[M * 32 + nC]      = f2bf(p0);       // C-layout -> LDS
      pb[M * 32 + 16 + nC] = f2bf(p1);
    }
    asm volatile("s_wait_dscnt 0" ::: "memory");   // same-wave LDS RAW

    // ---- O += P * V from LDS ----
    v16bf pf = load_a16(pb, 32, lane);
    o0 = WMMA_BF16(pf, load_b16(Vl,           32, lane), o0);
    o1 = WMMA_BF16(pf, load_b16(Vl + 16 * 32, 32, lane), o1);
    o2 = WMMA_BF16(pf, load_b16(Vl + 32 * 32, 32, lane), o2);
    o3 = WMMA_BF16(pf, load_b16(Vl + 48 * 32, 32, lane), o3);
    __syncthreads();                         // before next chunk overwrites K/V
  }

  // ---- epilogue: normalize, write Y (bf16, head-merged [s][h*64+d]) ----
#pragma unroll
  for (int r = 0; r < 8; ++r) {
    int M = r + 8 * hsel;
    float inv = 1.0f / l[r];
    size_t row = (size_t)(q0 + M) * DM + h * HD + nC;
    Yb[row]      = f2bf(o0[r] * inv);
    Yb[row + 16] = f2bf(o1[r] * inv);
    Yb[row + 32] = f2bf(o2[r] * inv);
    Yb[row + 48] = f2bf(o3[r] * inv);
  }
}

// ---------------------------------------------------------------------------
// Output projection: Y(bf16) @ Wp^T -> fp32 out.  Wave = 32x32 tile.
// ---------------------------------------------------------------------------
__global__ void __launch_bounds__(128) proj_kernel(const unsigned short* __restrict__ Yb,
                                                   const unsigned short* __restrict__ Wp,
                                                   float* __restrict__ out) {
  int lane = threadIdx.x & 31;
  int wid  = threadIdx.x >> 5;
  int tile = blockIdx.x * 4 + wid;           // 128 * 24 tiles
  int mt = tile / 24;
  int nt = tile % 24;
  int m0 = mt << 5;
  int n0 = nt << 5;

  const unsigned short* a0p = Yb + (size_t)m0 * DM;
  const unsigned short* a1p = a0p + (size_t)16 * DM;
  const unsigned short* b0p = Wp + (size_t)n0 * DM;
  const unsigned short* b1p = b0p + (size_t)16 * DM;

  v8f c00 = {}, c01 = {}, c10 = {}, c11 = {};
  for (int k = 0; k < DM; k += 32) {
    __builtin_prefetch(a0p + k + 32, 0, 3);
    v16bf a0 = load_a16(a0p + k, DM, lane);
    v16bf a1 = load_a16(a1p + k, DM, lane);
    v16bf b0 = load_b16(b0p + k, DM, lane);
    v16bf b1 = load_b16(b1p + k, DM, lane);
    c00 = WMMA_BF16(a0, b0, c00);
    c01 = WMMA_BF16(a0, b1, c01);
    c10 = WMMA_BF16(a1, b0, c10);
    c11 = WMMA_BF16(a1, b1, c11);
  }

  int hsel = lane >> 4;
  int nC   = lane & 15;
#pragma unroll
  for (int mi = 0; mi < 2; ++mi) {
#pragma unroll
    for (int ni = 0; ni < 2; ++ni) {
      v8f c = (mi == 0) ? ((ni == 0) ? c00 : c01) : ((ni == 0) ? c10 : c11);
#pragma unroll
      for (int r = 0; r < 8; ++r) {
        int mrow = m0 + mi * 16 + r + 8 * hsel;
        out[(size_t)mrow * DM + n0 + ni * 16 + nC] = c[r];
      }
    }
  }
}

// ---------------------------------------------------------------------------
// Host launcher
// ---------------------------------------------------------------------------
extern "C" void kernel_launch(void* const* d_in, const int* in_sizes, int n_in,
                              void* d_out, int out_size, void* d_ws, size_t ws_size,
                              hipStream_t stream) {
  (void)in_sizes; (void)n_in; (void)out_size; (void)ws_size;
  const float* x  = (const float*)d_in[0];
  const float* Wq = (const float*)d_in[1];
  const float* Wk = (const float*)d_in[2];
  const float* Wv = (const float*)d_in[3];
  const float* Wp = (const float*)d_in[4];

  unsigned short* ws = (unsigned short*)d_ws;
  size_t off = 0;
  unsigned short* xb  = ws + off; off += (size_t)SEQ * DM;
  unsigned short* wqb = ws + off; off += (size_t)DM * DM;
  unsigned short* wkb = ws + off; off += (size_t)DM * DM;
  unsigned short* wvb = ws + off; off += (size_t)DM * DM;
  unsigned short* wpb = ws + off; off += (size_t)DM * DM;
  unsigned short* Qb  = ws + off; off += (size_t)NH * SEQ * HD;  // [h][s][d]
  unsigned short* Kb  = ws + off; off += (size_t)NH * SEQ * HD;  // [h][s][d]
  unsigned short* Vt  = ws + off; off += (size_t)NH * HD * SEQ;  // [h][d][s]
  unsigned short* Yb  = ws + off; off += (size_t)SEQ * DM;       // [s][h*64+d]

  const int nX = SEQ * DM;
  const int nW = DM * DM;
  cvt_f32_bf16<<<(nX + 255) / 256, 256, 0, stream>>>(x,  xb,  nX);
  cvt_f32_bf16<<<(nW + 255) / 256, 256, 0, stream>>>(Wq, wqb, nW);
  cvt_f32_bf16<<<(nW + 255) / 256, 256, 0, stream>>>(Wk, wkb, nW);
  cvt_f32_bf16<<<(nW + 255) / 256, 256, 0, stream>>>(Wv, wvb, nW);
  cvt_f32_bf16<<<(nW + 255) / 256, 256, 0, stream>>>(Wp, wpb, nW);

  // 128*72 wave-tiles, 4 per 128-thread block
  qkv_kernel<<<(128 * 72) / 4, 128, 0, stream>>>(xb, wqb, wkb, wvb, Qb, Kb, Vt);

  // 12 heads * 64 query-blocks of 64 rows
  attn_kernel<<<NH * 64, 128, 0, stream>>>(Qb, Kb, Vt, Yb);

  // 128*24 wave-tiles, 4 per block
  proj_kernel<<<(128 * 24) / 4, 128, 0, stream>>>(Yb, wpb, (float*)d_out);
}
